// BipartiteGCNStack_38336878084420
// MI455X (gfx1250) — matrix operand order repaired and checked by reading, not compiled, not run on hardware
//
#include <hip/hip_runtime.h>
#include <stdint.h>

typedef unsigned short u16;
typedef unsigned int   u32;
typedef __bf16 bf16;
typedef bf16  v16bf __attribute__((ext_vector_type(16)));
typedef float v8f   __attribute__((ext_vector_type(8)));
typedef u16   v8u   __attribute__((ext_vector_type(8)));
typedef u16   v16u  __attribute__((ext_vector_type(16)));
typedef int   v4i   __attribute__((ext_vector_type(4)));

// address-space-qualified int4 pointers for the async global->LDS builtin
typedef __attribute__((address_space(1))) v4i* gv4p;   // global (AS1)
typedef __attribute__((address_space(3))) v4i* lv4p;   // LDS (AS3)

#if defined(__gfx1250__) && __has_builtin(__builtin_amdgcn_global_load_async_to_lds_b128) && __has_builtin(__builtin_amdgcn_s_wait_asynccnt)
#define USE_ASYNC_B 1
#else
#define USE_ASYNC_B 0
#endif

// ---------- helpers ----------
static __device__ __forceinline__ u16 f2bf(float f) {
  u32 u = __builtin_bit_cast(u32, f);
  u += 0x7fffu + ((u >> 16) & 1u);      // round-to-nearest-even
  return (u16)(u >> 16);
}
static __device__ __forceinline__ v16bf mk16(v8u lo, v8u hi) {
  v16u t;
#pragma unroll
  for (int i = 0; i < 8; ++i) { t[i] = lo[i]; t[i + 8] = hi[i]; }
  return __builtin_bit_cast(v16bf, t);
}
static __device__ __forceinline__ v8f zero8() {
  v8f z;
#pragma unroll
  for (int i = 0; i < 8; ++i) z[i] = 0.f;
  return z;
}

// ---------- WMMA bf16 GEMM:  C[M][N] = op(A)[M][K] * B[K][N] ----------
// A fp32 in global, converted to bf16 while staging into LDS (optionally
// scaled by 1/clamp(scaleRows,1e-8)); B given transposed as bf16 Bt[N][K],
// staged via async global->LDS copies when available.  Fused row/col sums
// of raw op(A).  Double-buffered LDS, one barrier per K-step.
// Block = 256 threads (8 waves): 4 m-tiles x 2 n-groups, BM=64, BK=32.
#define LDP 40   // padded LDS row stride (u16 units): 80B rows, 16B aligned

template<int NFR, bool TRANSA>
__global__ __launch_bounds__(256) void gemm_wmma(
    const float* __restrict__ Am, int lda,
    const u16*  __restrict__ Bt,
    float* __restrict__ C,
    float* __restrict__ sums,
    const float* __restrict__ scaleRows,
    const float* __restrict__ biasN,
    int M, int K)
{
  constexpr int N  = NFR * 32;   // 64 / 128 / 256
  constexpr int NB = N / 64;     // B rows per thread per tile
  __shared__ u16  lA[2][64 * LDP];
  __shared__ u16  lB[2][N * LDP];
  __shared__ float lsum[64];

  const int tid  = threadIdx.x;
  const int lane = tid & 31;
  const int wid  = tid >> 5;
  const int mt   = wid >> 1;          // m-tile (16 rows each)
  const int ng   = wid & 1;           // n-group (N/2 cols each)
  const int m0   = blockIdx.x * 64;

  // loader coordinates
  const int lr0 = tid >> 3, cgrp = (tid & 7) << 2;   // no-trans: row, 4-col group
  const int ti  = tid >> 4, j0   = (tid & 15) << 2;  // trans: k-row, 4 s-cols
  const int bn  = tid >> 2, bc   = (tid & 3) << 3;   // B: row, 8-elem group

  float rs0 = 0.f, rs1 = 0.f, cs0 = 0.f, cs1 = 0.f, cs2 = 0.f, cs3 = 0.f;
  float inv0 = 1.f, inv1 = 1.f;
  if (!TRANSA && scaleRows) {
    inv0 = 1.f / fmaxf(scaleRows[m0 + lr0     ], 1e-8f);
    inv1 = 1.f / fmaxf(scaleRows[m0 + lr0 + 32], 1e-8f);
  }
  if (tid < 64) lsum[tid] = 0.f;

  v8f acc[NFR];
#pragma unroll
  for (int f = 0; f < NFR; ++f) acc[f] = zero8();

  float4 va[2];
#if !USE_ASYNC_B
  v8u vb[NB];
#endif

  auto loadA = [&](int kb) {
    if (!TRANSA) {
      va[0] = *(const float4*)(Am + (size_t)(m0 + lr0     ) * lda + kb + cgrp);
      va[1] = *(const float4*)(Am + (size_t)(m0 + lr0 + 32) * lda + kb + cgrp);
    } else {
      va[0] = *(const float4*)(Am + (size_t)(kb + ti     ) * lda + m0 + j0);
      va[1] = *(const float4*)(Am + (size_t)(kb + ti + 16) * lda + m0 + j0);
    }
  };
  auto storeA = [&](int buf) {           // also accumulates raw-A sums
    if (!TRANSA) {
#pragma unroll
      for (int p = 0; p < 2; ++p) {
        const float4 v = va[p];
        const int r = lr0 + p * 32;
        const float inv = p ? inv1 : inv0;
        const float s4 = (v.x + v.y) + (v.z + v.w);
        if (p) rs1 += s4; else rs0 += s4;
        const u32 lo = (u32)f2bf(v.x * inv) | ((u32)f2bf(v.y * inv) << 16);
        const u32 hi = (u32)f2bf(v.z * inv) | ((u32)f2bf(v.w * inv) << 16);
        *(uint2*)&lA[buf][r * LDP + cgrp] = make_uint2(lo, hi);
      }
    } else {
#pragma unroll
      for (int p = 0; p < 2; ++p) {
        const float4 v = va[p];
        const int i = ti + p * 16;
        cs0 += v.x; cs1 += v.y; cs2 += v.z; cs3 += v.w;
        lA[buf][(j0 + 0) * LDP + i] = f2bf(v.x);
        lA[buf][(j0 + 1) * LDP + i] = f2bf(v.y);
        lA[buf][(j0 + 2) * LDP + i] = f2bf(v.z);
        lA[buf][(j0 + 3) * LDP + i] = f2bf(v.w);
      }
    }
  };

#if USE_ASYNC_B
  auto issueB = [&](int kb, int buf) {   // async global -> LDS (ASYNCcnt)
#pragma unroll
    for (int q = 0; q < NB; ++q) {
      const int n = bn + q * 64;
      __builtin_amdgcn_global_load_async_to_lds_b128(
          (gv4p)(Bt + (size_t)n * K + kb + bc),
          (lv4p)&lB[buf][n * LDP + bc], 0, 0);
    }
  };
#else
  auto loadB = [&](int kb) {
#pragma unroll
    for (int q = 0; q < NB; ++q)
      vb[q] = *(const v8u*)(Bt + (size_t)(bn + q * 64) * K + kb + bc);
  };
  auto storeB = [&](int buf) {
#pragma unroll
    for (int q = 0; q < NB; ++q)
      *(v8u*)&lB[buf][(bn + q * 64) * LDP + bc] = vb[q];
  };
#endif

  const int nb0 = ng * (N >> 1) + (lane & 15);
  auto compute = [&](int buf) {
    // A fragment (ISA 16-bit A 16x32): lanes 0-15 -> K 0-7,16-23;
    // lanes 16-31 -> K 8-15,24-31; M = lane%16 within this wave's m-tile.
    const u16* ap = &lA[buf][(mt * 16 + (lane & 15)) * LDP + ((lane >> 4) << 3)];
    const v16bf av = mk16(*(const v8u*)ap, *(const v8u*)(ap + 16));
#pragma unroll
    for (int f = 0; f < NFR; ++f) {
      // B fragment: lane = column n; lanes 0-15 K 0-7,16-23, lanes 16-31 K 8-15,24-31
      const u16* bp = &lB[buf][(nb0 + f * 16) * LDP + ((lane >> 4) << 3)];
      const v16bf bv = mk16(*(const v8u*)bp, *(const v8u*)(bp + 16));
      acc[f] = __builtin_amdgcn_wmma_f32_16x16x32_bf16(
          false, av, false, bv, (short)0, acc[f], false, false);
    }
  };

  // ---- software pipeline: double-buffered, one barrier per K-step ----
  const int nsteps = K >> 5;
  int cur = 0;

  loadA(0);
#if USE_ASYNC_B
  issueB(0, 0);
#else
  loadB(0);
  storeB(0);
#endif
  storeA(0);
#if USE_ASYNC_B
  __builtin_amdgcn_s_wait_asynccnt(0);
#endif
  __syncthreads();

  for (int it = 0; it < nsteps; ++it) {
    const bool more = (it + 1) < nsteps;
    const int  knxt = (it + 1) << 5;
    if (more) {                 // issue next tile's traffic first
      loadA(knxt);
#if USE_ASYNC_B
      issueB(knxt, cur ^ 1);
#else
      loadB(knxt);
#endif
    }
    compute(cur);               // WMMAs overlap outstanding loads
    if (more) {
      storeA(cur ^ 1);
#if !USE_ASYNC_B
      storeB(cur ^ 1);
#endif
    }
#if USE_ASYNC_B
    __builtin_amdgcn_s_wait_asynccnt(0);
#endif
    __syncthreads();
    cur ^= 1;
  }

  // ---- C writeback (VGPR r holds rows r / r+8 by lane half) ----
  const int mrow = m0 + mt * 16 + ((lane >> 4) << 3);
#pragma unroll
  for (int f = 0; f < NFR; ++f) {
    const int n = ng * (N >> 1) + f * 16 + (lane & 15);
    const float bv = biasN ? biasN[n] : 0.f;
#pragma unroll
    for (int r = 0; r < 8; ++r)
      C[(size_t)(mrow + r) * N + n] = acc[f][r] + bv;
  }

  if (sums) { // fused row/col sums of raw A (one owner workgroup per index)
    __syncthreads();
    if (!TRANSA) {
      atomicAdd(&lsum[lr0     ], rs0);
      atomicAdd(&lsum[lr0 + 32], rs1);
    } else {
      atomicAdd(&lsum[j0 + 0], cs0); atomicAdd(&lsum[j0 + 1], cs1);
      atomicAdd(&lsum[j0 + 2], cs2); atomicAdd(&lsum[j0 + 3], cs3);
    }
    __syncthreads();
    if (tid < 64) sums[m0 + tid] = lsum[tid];
  }
}

// ---------- transpose fp32 -> bf16, optional fused (x+bias-mean)*s+beta, ReLU ----------
__global__ void xform_transpose_bf16(
    const float* __restrict__ X, u16* __restrict__ XT, int R, int Ccols,
    const float* __restrict__ bias, const float* __restrict__ gamma,
    const float* __restrict__ beta, const float* __restrict__ mean,
    const float* __restrict__ var)
{
  __shared__ float tile[32][33];
  const int c0 = blockIdx.x * 32, r0 = blockIdx.y * 32;
  const int tx = threadIdx.x, ty = threadIdx.y;
  for (int i = ty; i < 32; i += 8) {
    const int r = r0 + i, c = c0 + tx;
    float v = 0.f;
    if (r < R && c < Ccols) {
      v = X[(size_t)r * Ccols + c];
      if (gamma) {
        const float s = gamma[c] * rsqrtf(var[c] + 1e-5f);
        v = (v + bias[c] - mean[c]) * s + beta[c];
        v = fmaxf(v, 0.f);
      }
    }
    tile[i][tx] = v;
  }
  __syncthreads();
  for (int i = ty; i < 32; i += 8) {
    const int c = c0 + i, r = r0 + tx;
    if (c < Ccols && r < R) XT[(size_t)c * R + r] = f2bf(tile[tx][i]);
  }
}

// ---------- elementwise BN(+bias)+ReLU, fp32 out ----------
__global__ void bn_relu_kernel(
    const float* __restrict__ Z, const float* __restrict__ bias,
    const float* __restrict__ gamma, const float* __restrict__ beta,
    const float* __restrict__ mean, const float* __restrict__ var,
    float* __restrict__ Y, int total, int Ccols)
{
  const int idx = blockIdx.x * 256 + threadIdx.x;
  if (idx >= total) return;
  const int j = idx - (idx / Ccols) * Ccols;
  const float s  = gamma[j] * rsqrtf(var[j] + 1e-5f);
  const float sh = beta[j] + (bias[j] - mean[j]) * s;
  const float v  = fmaf(Z[idx], s, sh);
  Y[idx] = v > 0.f ? v : 0.f;
}

// ---------- host ----------
static void launch_gemm(int N, bool transA,
                        const float* Am, int lda, const u16* Bt, float* C,
                        float* sums, const float* scaleRows, const float* biasN,
                        int M, int K, hipStream_t stream)
{
  const dim3 g(M / 64), b(256);
  if (transA) {                       // only N==128 used on the transposed path
    gemm_wmma<4, true ><<<g, b, 0, stream>>>(Am, lda, Bt, C, sums, scaleRows, biasN, M, K);
  } else if (N == 256) {
    gemm_wmma<8, false><<<g, b, 0, stream>>>(Am, lda, Bt, C, sums, scaleRows, biasN, M, K);
  } else if (N == 128) {
    gemm_wmma<4, false><<<g, b, 0, stream>>>(Am, lda, Bt, C, sums, scaleRows, biasN, M, K);
  } else {                            // N == 64
    gemm_wmma<2, false><<<g, b, 0, stream>>>(Am, lda, Bt, C, sums, scaleRows, biasN, M, K);
  }
}

extern "C" void kernel_launch(void* const* d_in, const int* in_sizes, int n_in,
                              void* d_out, int out_size, void* d_ws, size_t ws_size,
                              hipStream_t stream)
{
  (void)in_sizes; (void)n_in; (void)out_size; (void)ws_size;
  const float* Hs    = (const float*)d_in[0];
  /* d_in[1] H_target is unused by the reference */
  const float* A     = (const float*)d_in[2];
  const float* W0    = (const float*)d_in[3];
  const float* b0    = (const float*)d_in[4];
  const float* Wb0   = (const float*)d_in[5];
  const float* bb0   = (const float*)d_in[6];
  const float* W1    = (const float*)d_in[7];
  const float* b1    = (const float*)d_in[8];
  const float* bnf_g = (const float*)d_in[9];
  const float* bnf_b = (const float*)d_in[10];
  const float* bnf_m = (const float*)d_in[11];
  const float* bnf_v = (const float*)d_in[12];
  const float* bnb_g = (const float*)d_in[13];
  const float* bnb_b = (const float*)d_in[14];
  const float* bnb_m = (const float*)d_in[15];
  const float* bnb_v = (const float*)d_in[16];
  const float* Wout  = (const float*)d_in[17];
  const float* bout  = (const float*)d_in[18];

  const int NT = 16384, NS = 8192, DS = 256, DH = 128, DO = 64;

  // workspace layout (with reuse); ~46.4 MB total
  char* w = (char*)d_ws;
  const size_t oM1   = 0;                                  // M1 [NT][DS] f32; later M3 [NT][DH]
  const size_t oZ1   = oM1  + (size_t)NT * DS * 4;         // Z1 [NT][DH] f32; later Z3
  const size_t oT    = oZ1  + (size_t)NT * DH * 4;         // HsT (4MB) -> htT (4MB) -> hsT (2MB)
  const size_t oM2   = oT   + (size_t)DS * NS * 2;         // M2 [NS][DH] f32
  const size_t oZ2   = oM2  + (size_t)NS * DH * 4;         // Z2 [NS][DH] f32
  const size_t oH2   = oZ2  + (size_t)NS * DH * 4;         // H2 [NT][DH] f32
  const size_t oRS   = oH2  + (size_t)NT * DH * 4;         // rowsum [NT]
  const size_t oCS   = oRS  + (size_t)NT * 4;              // colsum [NS]
  const size_t oW0T  = oCS  + (size_t)NS * 4;
  const size_t oWb0T = oW0T + (size_t)DH * DS * 2;
  const size_t oW1T  = oWb0T+ (size_t)DH * DH * 2;
  const size_t oWoT  = oW1T + (size_t)DH * DH * 2;

  float* M1     = (float*)(w + oM1);
  float* M3     = (float*)(w + oM1);
  float* Z1     = (float*)(w + oZ1);
  float* Z3     = (float*)(w + oZ1);
  u16*   TbufT  = (u16*)  (w + oT);    // HsT / htT / hsT
  float* M2     = (float*)(w + oM2);
  float* Z2     = (float*)(w + oZ2);
  float* H2     = (float*)(w + oH2);
  float* rowsum = (float*)(w + oRS);
  float* colsum = (float*)(w + oCS);
  u16*   W0T    = (u16*)  (w + oW0T);
  u16*   Wb0T   = (u16*)  (w + oWb0T);
  u16*   W1T    = (u16*)  (w + oW1T);
  u16*   WoutT  = (u16*)  (w + oWoT);

  const dim3 tb(32, 8);
  const float* NF = nullptr;

  // 1) HsT = bf16(H_source^T)  [DS][NS]
  xform_transpose_bf16<<<dim3(DS/32, NS/32), tb, 0, stream>>>(Hs, TbufT, NS, DS, NF, NF, NF, NF, NF);
  // 2) M1 = A @ Hs (fp32 acc) + rowsum   (big GEMM #1, fused rowsum)
  launch_gemm(DS, false, A, NS, TbufT, M1, rowsum, nullptr, nullptr, NT, NS, stream);
  // 3) W0T
  xform_transpose_bf16<<<dim3(DH/32, DS/32), tb, 0, stream>>>(W0, W0T, DS, DH, NF, NF, NF, NF, NF);
  // 4) Z1 = (M1 / rowsum) @ W0
  launch_gemm(DH, false, M1, DS, W0T, Z1, nullptr, rowsum, nullptr, NT, DS, stream);
  // 5) htT = bf16( relu(bn0(Z1 + b0)) )^T   [DH][NT]
  xform_transpose_bf16<<<dim3(DH/32, NT/32), tb, 0, stream>>>(Z1, TbufT, NT, DH, b0, bnf_g, bnf_b, bnf_m, bnf_v);
  // 6) M2 = A^T @ h_tgt + colsum        (big GEMM #2, transposed-A, fused colsum)
  launch_gemm(DH, true, A, NS, TbufT, M2, colsum, nullptr, nullptr, NS, NT, stream);
  // 7) Wb0T
  xform_transpose_bf16<<<dim3(DH/32, DH/32), tb, 0, stream>>>(Wb0, Wb0T, DH, DH, NF, NF, NF, NF, NF);
  // 8) Z2 = (M2 / colsum) @ Wb0
  launch_gemm(DH, false, M2, DH, Wb0T, Z2, nullptr, colsum, nullptr, NS, DH, stream);
  // 9) hsT = bf16( relu(bn_b(Z2 + bb0)) )^T  [DH][NS]
  xform_transpose_bf16<<<dim3(DH/32, NS/32), tb, 0, stream>>>(Z2, TbufT, NS, DH, bb0, bnb_g, bnb_b, bnb_m, bnb_v);
  // 10) M3 = A @ h_src                   (big GEMM #3, rowsum already known)
  launch_gemm(DH, false, A, NS, TbufT, M3, nullptr, nullptr, nullptr, NT, NS, stream);
  // 11) W1T
  xform_transpose_bf16<<<dim3(DH/32, DH/32), tb, 0, stream>>>(W1, W1T, DH, DH, NF, NF, NF, NF, NF);
  // 12) Z3 = (M3 / rowsum) @ W1
  launch_gemm(DH, false, M3, DH, W1T, Z3, nullptr, rowsum, nullptr, NT, DH, stream);
  // 13) H2 = relu(bn1(Z3 + b1))  (fp32, row-major; feeds final GEMM as A)
  bn_relu_kernel<<<(NT*DH + 255)/256, 256, 0, stream>>>(Z3, b1, bnf_g + DH, bnf_b + DH, bnf_m + DH, bnf_v + DH, H2, NT*DH, DH);
  // 14) WoutT
  xform_transpose_bf16<<<dim3(DO/32, DH/32), tb, 0, stream>>>(Wout, WoutT, DH, DO, NF, NF, NF, NF, NF);
  // 15) out = H2 @ Wout + bout  -> d_out [NT][DO] fp32
  launch_gemm(DO, false, H2, DH, WoutT, (float*)d_out, nullptr, nullptr, bout, NT, DH, stream);
}
